// Transformer_36764920054093
// MI455X (gfx1250) — matrix operand order repaired and checked by reading, not compiled, not run on hardware
//
#include <hip/hip_runtime.h>

// ---------------- model dimensions (compile-time constants) ----------------
#define B_   8
#define S_   512          // src len == trg len
#define E_   512          // embed
#define H_   8            // heads
#define HD_  64           // head dim
#define FF_  2048         // ffn hidden
#define V_   32000        // vocab
#define L_   6            // layers

typedef __attribute__((ext_vector_type(16))) __bf16        v16bf;
typedef __attribute__((ext_vector_type(8)))  float         v8f;
typedef __attribute__((ext_vector_type(4)))  unsigned int  u32x4;
typedef __attribute__((ext_vector_type(4)))  float         f32x4;

union FragU {
  u32x4 q[2];   // 32 bytes = 16 bf16
  v16bf v;
};

__device__ __forceinline__ unsigned short f2bf(float f) {
  unsigned int u = __float_as_uint(f);
  u += 0x7FFFu + ((u >> 16) & 1u);          // round-to-nearest-even
  return (unsigned short)(u >> 16);
}
__device__ __forceinline__ float bf2f(unsigned short h) {
  return __uint_as_float(((unsigned int)h) << 16);
}

// CDNA5 async global->LDS DMA (ASYNCcnt tracked); lds_off = byte offset in
// the workgroup LDS allocation, gaddr = 64-bit global address.
__device__ __forceinline__ void async_ld_b128(unsigned lds_off, const void* gaddr) {
  asm volatile("global_load_async_to_lds_b128 %0, %1, off"
               :: "v"(lds_off), "v"((unsigned long long)gaddr)
               : "memory");
}
__device__ __forceinline__ void wait_async0() {
  asm volatile("s_wait_asynccnt 0x0" ::: "memory");
}

__device__ __forceinline__ float wave_max(float v) {
#pragma unroll
  for (int o = 16; o > 0; o >>= 1) v = fmaxf(v, __shfl_xor(v, o, 32));
  return v;
}
__device__ __forceinline__ float wave_sum(float v) {
#pragma unroll
  for (int o = 16; o > 0; o >>= 1) v += __shfl_xor(v, o, 32);
  return v;
}

// ---------------------------------------------------------------------------
// f32 -> bf16 bulk conversion (8 elements / thread, count % 2048 == 0)
// ---------------------------------------------------------------------------
__global__ __launch_bounds__(256)
void tf_cvt_bf16(const float* __restrict__ in, unsigned short* __restrict__ out)
{
  const long i = ((long)blockIdx.x * 256 + threadIdx.x) * 8;
  f32x4 a = *(const f32x4*)&in[i];
  f32x4 b = *(const f32x4*)&in[i + 4];
  u32x4 o;
  o.x = (unsigned)f2bf(a.x) | ((unsigned)f2bf(a.y) << 16);
  o.y = (unsigned)f2bf(a.z) | ((unsigned)f2bf(a.w) << 16);
  o.z = (unsigned)f2bf(b.x) | ((unsigned)f2bf(b.y) << 16);
  o.w = (unsigned)f2bf(b.z) | ((unsigned)f2bf(b.w) << 16);
  *(u32x4*)&out[i] = o;
}

// ---------------------------------------------------------------------------
// WMMA GEMM on bf16 operands:  C = act( A(MxK) * W^T + bias + residual )
//   flags bit0: add bias[n] (f32)      bit1: add residual (bf16, C layout)
//   flags bit2: relu                   bit3: BN mode (W stored K x N)
//   flags bit4: C is f32 (else bf16)
// Batched via gridDim.z: z -> (b = z/nH, h = z%nH), offsets in elements.
// Block: 256 thr = 8 waves (4M x 2N), wave owns 32x32 patch = 2x2 WMMA tiles.
// Block tile 128(M) x 64(N), K-step 64, double-buffered LDS: async DMA for
// tile t+1 is issued right after the barrier, overlapping the 8 WMMAs of
// tile t.  M%128==0, N%64==0, K%64==0 (true for every GEMM here).
// ---------------------------------------------------------------------------
#define TM 128
#define TN 64
#define TK 64
#define LDSW 72   // halfword row stride (144B: 16B-aligned)

#define FB_BIAS   1
#define FB_RES    2
#define FB_RELU   4
#define FB_BN     8
#define FB_F32OUT 16

__global__ __launch_bounds__(256)
void tf_gemm(const unsigned short* __restrict__ A,
             const unsigned short* __restrict__ Wm,
             void* __restrict__ C, const float* __restrict__ bias,
             const unsigned short* __restrict__ Res,
             int K, int lda, int ldw, int ldc, int nH,
             long sAb, long sAh, long sWb, long sWh, long sCb, long sCh,
             int flags)
{
  __shared__ unsigned short As[2][TM * LDSW];   // 2 x (128 x 64) bf16
  __shared__ unsigned short Bs[2][TN * LDSW];   // 2 x ( 64 x 64) bf16 Bt[n][k]

  const int tid  = threadIdx.x;
  const int lane = tid & 31;
  const int wave = tid >> 5;
  const int wm   = wave >> 1;    // 4 M-waves
  const int wn   = wave & 1;     // 2 N-waves

  const int z  = blockIdx.z;
  const int bz = z / nH;
  const int hz = z - bz * nH;

  const unsigned short* Ab = A  + (long)bz * sAb + (long)hz * sAh;
  const unsigned short* Wb = Wm + (long)bz * sWb + (long)hz * sWh;
  const long coff = (long)bz * sCb + (long)hz * sCh;

  const int m0 = blockIdx.y * TM;
  const int n0 = blockIdx.x * TN;

  // LDS byte offsets (low 32 bits of generic address)
  unsigned asBase[2], bsBase[2];
  asBase[0] = (unsigned)(unsigned long long)(void*)&As[0][0];
  asBase[1] = (unsigned)(unsigned long long)(void*)&As[1][0];
  bsBase[0] = (unsigned)(unsigned long long)(void*)&Bs[0][0];
  bsBase[1] = (unsigned)(unsigned long long)(void*)&Bs[1][0];

  // stage one 128x64 A tile + 64x64 B tile into LDS buffer `bi`
  auto stageTile = [&](int bi, int k0) {
    // A: 1024 x 16B chunks, 4 async DMAs / thread
#pragma unroll
    for (int i = 0; i < 4; ++i) {
      const int idx = tid + i * 256;                // 0..1023
      const int r = idx >> 3, c8 = (idx & 7) << 3;  // 8 chunks per row
      async_ld_b128(asBase[bi] + (unsigned)(r * (LDSW * 2) + c8 * 2),
                    &Ab[(long)(m0 + r) * lda + (k0 + c8)]);
    }
    if (flags & FB_BN) {
      // W stored K x N: two 16B loads along N, then LDS transpose stores
      const int kk0 = tid >> 3,        n80 = (tid & 7) << 3;
      const int kk1 = (tid + 256) >> 3, n81 = (tid & 7) << 3;
      union { u32x4 v; unsigned short us[8]; } t0, t1;
      t0.v = *(const u32x4*)&Wb[(long)(k0 + kk0) * ldw + (n0 + n80)];
      t1.v = *(const u32x4*)&Wb[(long)(k0 + kk1) * ldw + (n0 + n81)];
      unsigned short* bsp = &Bs[bi][0];
#pragma unroll
      for (int j = 0; j < 8; ++j) bsp[(n80 + j) * LDSW + kk0] = t0.us[j];
#pragma unroll
      for (int j = 0; j < 8; ++j) bsp[(n81 + j) * LDSW + kk1] = t1.us[j];
    } else {
      // W stored N x K: 512 x 16B chunks, 2 async DMAs / thread
#pragma unroll
      for (int i = 0; i < 2; ++i) {
        const int idx = tid + i * 256;                // 0..511
        const int r = idx >> 3, c8 = (idx & 7) << 3;
        async_ld_b128(bsBase[bi] + (unsigned)(r * (LDSW * 2) + c8 * 2),
                      &Wb[(long)(n0 + r) * ldw + (k0 + c8)]);
      }
    }
  };

  v8f acc00 = {}, acc01 = {}, acc10 = {}, acc11 = {};

  const int half = lane >> 4;
  const int l15  = lane & 15;
  const int am0  = wm * 32 + l15;
  const int am1  = wm * 32 + 16 + l15;
  const int bn0  = wn * 32 + l15;
  const int bn1  = wn * 32 + 16 + l15;

  stageTile(0, 0);
  int buf = 0;
  for (int k0 = 0; k0 < K; k0 += TK) {
    wait_async0();        // this wave's DMAs for tile `buf` have landed
    __syncthreads();      // ...and everyone else's; prior reads all done
    if (k0 + TK < K) stageTile(buf ^ 1, k0 + TK);   // overlap with WMMAs

    const unsigned short* Ap = &As[buf][0];
    const unsigned short* Bp = &Bs[buf][0];
#pragma unroll
    for (int ks = 0; ks < TK; ks += 32) {
      FragU fa0, fa1, fb0, fb1;
      // A: VGPR0-3 <- K=ks+8h+0..7 ; VGPR4-7 <- K=ks+16+8h+0..7
      fa0.q[0] = *(const u32x4*)&Ap[am0 * LDSW + ks + 8 * half];
      fa0.q[1] = *(const u32x4*)&Ap[am0 * LDSW + ks + 16 + 8 * half];
      fa1.q[0] = *(const u32x4*)&Ap[am1 * LDSW + ks + 8 * half];
      fa1.q[1] = *(const u32x4*)&Ap[am1 * LDSW + ks + 16 + 8 * half];
      // B: lane = column n, VGPR0-3 <- K=ks+8h+0..7 ; VGPR4-7 <- K=ks+16+8h+0..7
      const unsigned short* pb0 = &Bp[bn0 * LDSW + ks + 8 * half];
      const unsigned short* pb1 = &Bp[bn1 * LDSW + ks + 8 * half];
      fb0.q[0] = *(const u32x4*)pb0; fb0.q[1] = *(const u32x4*)(pb0 + 16);
      fb1.q[0] = *(const u32x4*)pb1; fb1.q[1] = *(const u32x4*)(pb1 + 16);

      acc00 = __builtin_amdgcn_wmma_f32_16x16x32_bf16(
          false, fa0.v, false, fb0.v, (short)0, acc00, false, false);
      acc01 = __builtin_amdgcn_wmma_f32_16x16x32_bf16(
          false, fa0.v, false, fb1.v, (short)0, acc01, false, false);
      acc10 = __builtin_amdgcn_wmma_f32_16x16x32_bf16(
          false, fa1.v, false, fb0.v, (short)0, acc10, false, false);
      acc11 = __builtin_amdgcn_wmma_f32_16x16x32_bf16(
          false, fa1.v, false, fb1.v, (short)0, acc11, false, false);
    }
    buf ^= 1;
  }

  // ---- epilogue: C tile layout: VGPR r -> M = 8*(lane>>4)+r, N = lane&15 ---
  const int mh8 = (lane >> 4) * 8;
#pragma unroll
  for (int in = 0; in < 2; ++in) {
    const int nn = n0 + wn * 32 + in * 16 + l15;
    const float bv = (flags & FB_BIAS) ? bias[nn] : 0.0f;
#pragma unroll
    for (int im = 0; im < 2; ++im) {
      const v8f* accp = (im == 0) ? ((in == 0) ? &acc00 : &acc01)
                                  : ((in == 0) ? &acc10 : &acc11);
      const int mb = m0 + wm * 32 + im * 16 + mh8;
#pragma unroll
      for (int r = 0; r < 8; ++r) {
        const long mm = mb + r;
        float vv = (*accp)[r] + bv;
        if (flags & FB_RES) vv += bf2f(Res[coff + mm * ldc + nn]);
        if (flags & FB_RELU) vv = fmaxf(vv, 0.0f);
        if (flags & FB_F32OUT) ((float*)C)[coff + mm * ldc + nn] = vv;
        else ((unsigned short*)C)[coff + mm * ldc + nn] = f2bf(vv);
      }
    }
  }
}

// ------------------- embedding + positional encoding (bf16 out) -------------
__global__ void tf_embed(const int* __restrict__ tok,
                         const float* __restrict__ emb,
                         unsigned short* __restrict__ out)
{
  const int row = blockIdx.x;              // b*S + s
  const int s   = row & (S_ - 1);
  const int t   = tok[row];
  const float* e = emb + (long)t * E_;
  unsigned short* o = out + (long)row * E_;
  for (int c = threadIdx.x; c < E_; c += blockDim.x) {
    const int   i2  = c & ~1;
    const float div = __expf(-(float)i2 * 0.017988946039016f); // ln(1e4)/512
    const float ang = (float)s * div;
    const float pe  = (c & 1) ? __cosf(ang) : __sinf(ang);
    o[c] = f2bf(e[c] + pe);
  }
}

// ---- masked, scaled softmax: f32 scores in, bf16 probabilities out ---------
__global__ __launch_bounds__(256)
void tf_softmax(const float* __restrict__ Sc, unsigned short* __restrict__ P,
                const int* __restrict__ tok, int causal)
{
  const int  wave = threadIdx.x >> 5, lane = threadIdx.x & 31;
  const long row  = (long)blockIdx.x * 8 + wave;   // (b*H + h)*S + q
  const int  q    = (int)(row & (S_ - 1));
  const long bh   = row >> 9;                      // / S_
  const int  b    = (int)(bh >> 3);                // / H_
  const float*     s = Sc + row * S_;
  unsigned short*  p = P  + row * S_;
  const int*      tb = tok + (long)b * S_;
  const float scale = 0.04419417382415922f;        // 1/sqrt(512)

  float vals[16];
  float m = -3.0e38f;
#pragma unroll
  for (int j = 0; j < 16; ++j) {
    const int k = j * 32 + lane;
    const bool valid = (tb[k] != 0) && (!causal || (k <= q));
    const float x = valid ? s[k] * scale : -1.0e30f;
    vals[j] = x;
    m = fmaxf(m, x);
  }
  m = wave_max(m);
  float sum = 0.0f;
#pragma unroll
  for (int j = 0; j < 16; ++j) { vals[j] = __expf(vals[j] - m); sum += vals[j]; }
  sum = wave_sum(sum);
  const float inv = 1.0f / sum;
#pragma unroll
  for (int j = 0; j < 16; ++j) p[j * 32 + lane] = f2bf(vals[j] * inv);
}

// ------------- layernorm: f32 in, bf16 out (wave per row of 512) ------------
__global__ __launch_bounds__(256)
void tf_layernorm(const float* __restrict__ X, unsigned short* __restrict__ O,
                  const float* __restrict__ g, const float* __restrict__ bta)
{
  const int  wave = threadIdx.x >> 5, lane = threadIdx.x & 31;
  const long row  = (long)blockIdx.x * 8 + wave;
  const float*    x = X + row * E_;
  unsigned short* o = O + row * E_;
  float vals[16];
  float s = 0.0f;
#pragma unroll
  for (int j = 0; j < 16; ++j) { vals[j] = x[j * 32 + lane]; s += vals[j]; }
  s = wave_sum(s);
  const float mean = s * (1.0f / (float)E_);
  float vsum = 0.0f;
#pragma unroll
  for (int j = 0; j < 16; ++j) { const float d = vals[j] - mean; vsum += d * d; }
  vsum = wave_sum(vsum) * (1.0f / (float)E_);
  const float inv = rsqrtf(vsum + 1e-5f);
#pragma unroll
  for (int j = 0; j < 16; ++j) {
    const int c = j * 32 + lane;
    o[c] = f2bf((vals[j] - mean) * inv * g[c] + bta[c]);
  }
}

// ---------------------------------------------------------------------------
static inline void gemm(hipStream_t st, const unsigned short* A,
                        const unsigned short* W, void* C,
                        const float* bias, const unsigned short* res,
                        int M, int N, int K, int lda, int ldw, int ldc,
                        int batch, int nH,
                        long sAb, long sAh, long sWb, long sWh, long sCb, long sCh,
                        int flags)
{
  dim3 grid(N / TN, M / TM, batch);
  tf_gemm<<<grid, 256, 0, st>>>(A, W, C, bias, res, K, lda, ldw, ldc, nH,
                                sAb, sAh, sWb, sWh, sCb, sCh, flags);
}
static inline void cvt(hipStream_t st, const float* in, unsigned short* out, long n)
{
  tf_cvt_bf16<<<(unsigned)(n / 2048), 256, 0, st>>>(in, out);
}

extern "C" void kernel_launch(void* const* d_in, const int* in_sizes, int n_in,
                              void* d_out, int out_size, void* d_ws, size_t ws_size,
                              hipStream_t stream)
{
  (void)in_sizes; (void)n_in; (void)out_size; (void)ws_size;

  const int*   src      = (const int*)  d_in[0];
  const int*   trg      = (const int*)  d_in[1];
  const float* src_emb  = (const float*)d_in[2];
  const float* trg_emb  = (const float*)d_in[3];
  const float* fc_w     = (const float*)d_in[4];
  const float* fc_b     = (const float*)d_in[5];
  const float* e_wq     = (const float*)d_in[6];
  const float* e_wk     = (const float*)d_in[7];
  const float* e_wv     = (const float*)d_in[8];
  const float* e_wo     = (const float*)d_in[9];
  const float* e_bo     = (const float*)d_in[10];
  const float* e_w1     = (const float*)d_in[11];
  const float* e_b1     = (const float*)d_in[12];
  const float* e_w2     = (const float*)d_in[13];
  const float* e_b2     = (const float*)d_in[14];
  const float* e_ln1g   = (const float*)d_in[15];
  const float* e_ln1b   = (const float*)d_in[16];
  const float* e_ln2g   = (const float*)d_in[17];
  const float* e_ln2b   = (const float*)d_in[18];
  const float* d_wq     = (const float*)d_in[19];
  const float* d_wk     = (const float*)d_in[20];
  const float* d_wv     = (const float*)d_in[21];
  const float* d_wo     = (const float*)d_in[22];
  const float* d_bo     = (const float*)d_in[23];
  const float* c_wq     = (const float*)d_in[24];
  const float* c_wk     = (const float*)d_in[25];
  const float* c_wv     = (const float*)d_in[26];
  const float* c_wo     = (const float*)d_in[27];
  const float* c_bo     = (const float*)d_in[28];
  const float* d_w1     = (const float*)d_in[29];
  const float* d_b1     = (const float*)d_in[30];
  const float* d_w2     = (const float*)d_in[31];
  const float* d_b2     = (const float*)d_in[32];
  const float* d_ln1g   = (const float*)d_in[33];
  const float* d_ln1b   = (const float*)d_in[34];
  const float* d_ln2g   = (const float*)d_in[35];
  const float* d_ln2b   = (const float*)d_in[36];
  const float* d_ln3g   = (const float*)d_in[37];
  const float* d_ln3b   = (const float*)d_in[38];

  // -------- workspace layout (bytes, 256B aligned segments) --------
  char* wp = (char*)d_ws;
  auto alloc = [&](size_t bytes) -> void* {
    void* r = wp; wp += (bytes + 255) & ~(size_t)255; return r;
  };
  typedef unsigned short us;
  const long HDW = (long)L_ * HD_ * HD_;      // 24576
  const long EEW = (long)L_ * E_ * E_;        // 1,572,864
  const long FFW = (long)L_ * FF_ * E_;       // 6,291,456
  const long BSE = (long)B_ * S_ * E_;        // 2,097,152
  const long SCN = (long)B_ * H_ * S_ * S_;   // 16,777,216

  // bf16 weights
  us* Ewq = (us*)alloc(HDW * 2);  us* Ewk = (us*)alloc(HDW * 2);
  us* Ewv = (us*)alloc(HDW * 2);  us* Ewo = (us*)alloc(EEW * 2);
  us* Ew1 = (us*)alloc(FFW * 2);  us* Ew2 = (us*)alloc(FFW * 2);
  us* Dwq = (us*)alloc(HDW * 2);  us* Dwk = (us*)alloc(HDW * 2);
  us* Dwv = (us*)alloc(HDW * 2);  us* Dwo = (us*)alloc(EEW * 2);
  us* Cwq = (us*)alloc(HDW * 2);  us* Cwk = (us*)alloc(HDW * 2);
  us* Cwv = (us*)alloc(HDW * 2);  us* Cwo = (us*)alloc(EEW * 2);
  us* Dw1 = (us*)alloc(FFW * 2);  us* Dw2 = (us*)alloc(FFW * 2);
  us* Fcw = (us*)alloc((long)V_ * E_ * 2);
  // bf16 activations
  us* encX = (us*)alloc(BSE * 2);   // also final enc_out
  us* decX = (us*)alloc(BSE * 2);
  us* Qb   = (us*)alloc(BSE * 2);
  us* Kb   = (us*)alloc(BSE * 2);
  us* Vb   = (us*)alloc(BSE * 2);
  us* Ob   = (us*)alloc(BSE * 2);
  us* Hb   = (us*)alloc((long)B_ * S_ * FF_ * 2);
  us* Pb   = (us*)alloc(SCN * 2);   // bf16 attention probabilities
  // f32 buffers
  float* Sc = (float*)alloc(SCN * 4);   // raw scores
  float* T1 = (float*)alloc(BSE * 4);   // pre-layernorm sums

  // -------- one-time (per call) weight conversion to bf16 --------
  cvt(stream, e_wq, Ewq, HDW);  cvt(stream, e_wk, Ewk, HDW);
  cvt(stream, e_wv, Ewv, HDW);  cvt(stream, e_wo, Ewo, EEW);
  cvt(stream, e_w1, Ew1, FFW);  cvt(stream, e_w2, Ew2, FFW);
  cvt(stream, d_wq, Dwq, HDW);  cvt(stream, d_wk, Dwk, HDW);
  cvt(stream, d_wv, Dwv, HDW);  cvt(stream, d_wo, Dwo, EEW);
  cvt(stream, c_wq, Cwq, HDW);  cvt(stream, c_wk, Cwk, HDW);
  cvt(stream, c_wv, Cwv, HDW);  cvt(stream, c_wo, Cwo, EEW);
  cvt(stream, d_w1, Dw1, FFW);  cvt(stream, d_w2, Dw2, FFW);
  cvt(stream, fc_w, Fcw, (long)V_ * E_);

  const int rowsBS   = B_ * S_;               // 4096
  const int rowsBSH  = B_ * S_ * H_;          // 32768
  const int lnBlocks = rowsBS / 8;            // 512
  const int smBlocks = (B_ * H_ * S_) / 8;    // 4096

  const long sQ_b = (long)S_ * E_;   // per-batch stride in Q/K/V/O buffers
  const long sQ_h = HD_;
  const long sS_b = (long)H_ * S_ * S_;
  const long sS_h = (long)S_ * S_;

  // per-head projection: x(B*S*H,64) @ W(64,64)^T  -> bf16 out
  auto proj = [&](const us* x, const us* w, us* out) {
    gemm(stream, x, w, out, nullptr, nullptr,
         rowsBSH, HD_, HD_, HD_, HD_, HD_, 1, 1, 0, 0, 0, 0, 0, 0, 0);
  };
  auto attention = [&](us* x_q, const us* x_kv,
                       const us* wq, const us* wk, const us* wv,
                       const us* wo, const float* bo,
                       const int* mtok, int causal,
                       const float* lng, const float* lnb) {
    proj(x_q,  wq, Qb);
    proj(x_kv, wk, Kb);
    proj(x_kv, wv, Vb);
    // scores[b,h] = Q @ K^T (f32 out, 64-way batch)
    gemm(stream, Qb, Kb, Sc, nullptr, nullptr,
         S_, S_, HD_, E_, E_, S_, B_ * H_, H_,
         sQ_b, sQ_h, sQ_b, sQ_h, sS_b, sS_h, FB_F32OUT);
    tf_softmax<<<smBlocks, 256, 0, stream>>>(Sc, Pb, mtok, causal);
    // out[b,:,h,:] = probs @ V  (BN mode: V is KxN, bf16 out)
    gemm(stream, Pb, Vb, Ob, nullptr, nullptr,
         S_, HD_, S_, S_, E_, E_, B_ * H_, H_,
         sS_b, sS_h, sQ_b, sQ_h, sQ_b, sQ_h, FB_BN);
    // wo projection + bias + residual(x_q) -> f32 T1, then LN -> bf16 x_q
    gemm(stream, Ob, wo, T1, bo, x_q,
         rowsBS, E_, E_, E_, E_, E_, 1, 1, 0, 0, 0, 0, 0, 0,
         FB_BIAS | FB_RES | FB_F32OUT);
    tf_layernorm<<<lnBlocks, 256, 0, stream>>>(T1, x_q, lng, lnb);
  };
  auto ffn = [&](us* x, const us* w1, const float* b1,
                 const us* w2, const float* b2,
                 const float* lng, const float* lnb) {
    gemm(stream, x, w1, Hb, b1, nullptr,
         rowsBS, FF_, E_, E_, E_, FF_, 1, 1, 0, 0, 0, 0, 0, 0,
         FB_BIAS | FB_RELU);
    gemm(stream, Hb, w2, T1, b2, x,
         rowsBS, E_, FF_, FF_, FF_, E_, 1, 1, 0, 0, 0, 0, 0, 0,
         FB_BIAS | FB_RES | FB_F32OUT);
    tf_layernorm<<<lnBlocks, 256, 0, stream>>>(T1, x, lng, lnb);
  };

  // -------- embeddings + positional encodings --------
  tf_embed<<<rowsBS, 256, 0, stream>>>(src, src_emb, encX);
  tf_embed<<<rowsBS, 256, 0, stream>>>(trg, trg_emb, decX);

  // -------- encoder --------
  for (int i = 0; i < L_; ++i) {
    const long wo_i = (long)i * E_ * E_, hd_i = (long)i * HD_ * HD_;
    attention(encX, encX,
              Ewq + hd_i, Ewk + hd_i, Ewv + hd_i,
              Ewo + wo_i, e_bo + (long)i * E_,
              src, 0, e_ln1g + (long)i * E_, e_ln1b + (long)i * E_);
    ffn(encX, Ew1 + (long)i * FF_ * E_, e_b1 + (long)i * FF_,
        Ew2 + (long)i * E_ * FF_, e_b2 + (long)i * E_,
        e_ln2g + (long)i * E_, e_ln2b + (long)i * E_);
  }

  // -------- decoder --------
  for (int i = 0; i < L_; ++i) {
    const long wo_i = (long)i * E_ * E_, hd_i = (long)i * HD_ * HD_;
    attention(decX, decX,
              Dwq + hd_i, Dwk + hd_i, Dwv + hd_i,
              Dwo + wo_i, d_bo + (long)i * E_,
              trg, 1, d_ln1g + (long)i * E_, d_ln1b + (long)i * E_);
    attention(decX, encX,
              Cwq + hd_i, Cwk + hd_i, Cwv + hd_i,
              Cwo + wo_i, c_bo + (long)i * E_,
              src, 0, d_ln2g + (long)i * E_, d_ln2b + (long)i * E_);
    ffn(decX, Dw1 + (long)i * FF_ * E_, d_b1 + (long)i * FF_,
        Dw2 + (long)i * E_ * FF_, d_b2 + (long)i * E_,
        d_ln3g + (long)i * E_, d_ln3b + (long)i * E_);
  }

  // -------- final projection to vocab (f32 logits to d_out) --------
  gemm(stream, decX, Fcw, d_out, fc_b, nullptr,
       rowsBS, V_, E_, E_, E_, V_, 1, 1, 0, 0, 0, 0, 0, 0,
       FB_BIAS | FB_F32OUT);
}
